// Corrformer_12970801234578
// MI455X (gfx1250) — compile-verified, matrix operand order checked by reading, not run on hardware
//
#include <hip/hip_runtime.h>
#include <hip/hip_bf16.h>
#include <math.h>

// ---------------------------------------------------------------------------
// Types / constants
// ---------------------------------------------------------------------------
typedef __attribute__((ext_vector_type(16))) _Float16 v16h;
typedef __attribute__((ext_vector_type(8)))  _Float16 v8h;
typedef __attribute__((ext_vector_type(8)))  float    v8f;

#define DM      256          // d_model
#define LSEQ    48           // enc seq len == dec len (LABEL+PRED)
#define NODES   32
#define BB      16           // batch
#define BN512   512          // BB*NODES
#define NH      8
#define HD      32           // head dim
#define DFF     1024
#define TCNC    1536         // HD*LSEQ
#define TCNK    4608         // 3*TCNC
#define TCNM    4096         // BB*NH*NODES
#define MROW    24576L       // BN512*LSEQ
#define TOPK    3
#define PREDL   24

// ---------------------------------------------------------------------------
// Optional async global->LDS path (probe builtins; fall back to ld/st)
// ---------------------------------------------------------------------------
#ifndef HAVE_ASYNC_LDS
#if defined(__has_builtin)
#if __has_builtin(__builtin_amdgcn_global_load_async_to_lds_b128) && \
    __has_builtin(__builtin_amdgcn_s_wait_asynccnt)
#define HAVE_ASYNC_LDS 1
#endif
#endif
#endif
#ifndef HAVE_ASYNC_LDS
#define HAVE_ASYNC_LDS 0
#endif

#if HAVE_ASYNC_LDS
// builtin signature (from clang diagnostic): (v4i AS1*, v4i AS3*, Ii, Ii)
typedef int v4i_t __attribute__((vector_size(16)));
typedef __attribute__((address_space(1))) v4i_t* gp1_t;
typedef __attribute__((address_space(3))) v4i_t* lp3_t;
#endif

// 16-byte global -> LDS copy (async when available)
__device__ __forceinline__ void lds_copy16(const _Float16* g, _Float16* l) {
#if HAVE_ASYNC_LDS
  gp1_t gp = (gp1_t)(unsigned long long)(size_t)g;
  lp3_t lp = (lp3_t)(unsigned)(size_t)l;   // low 32 bits of generic = LDS offset
  __builtin_amdgcn_global_load_async_to_lds_b128(gp, lp, 0, 0);
#else
  *(v8h*)l = *(const v8h*)g;
#endif
}

__device__ __forceinline__ void lds_copy_wait() {
#if HAVE_ASYNC_LDS
  __builtin_amdgcn_s_wait_asynccnt(0);
#endif
}

// ---------------------------------------------------------------------------
// WMMA GEMM:  C[M,N] = act( A(f16)[M,K] @ WT(f16)[N,K]^T + bias + addsrc )
//   weights are pre-transposed (N-major) so both LDS stages are 16B copies.
//   act: 0 = none, 1 = relu, 2 = gelu(exact)
//   block tile 64x128, K-step 32, 8 wave32 waves (4 M x 2 N), 4 WMMA/wave/step
// ---------------------------------------------------------------------------
#define BMT 64
#define BNT 128
#define BKT 32
#define LDK 40   // padded LDS row (halfs); 80B row keeps 16B slot alignment

// build a 16x32 f16 fragment per ISA layout: lane lr = row, halfs[0..7] =
// K kb..kb+7, halfs[8..15] = K kb+16..kb+23   (kb = (lane>>4)*8)
__device__ __forceinline__ v16h load_frag(const _Float16* row, int kb) {
  const v8h lo = *(const v8h*)(row + kb);
  const v8h hh = *(const v8h*)(row + kb + 16);
  v16h f;
#pragma unroll
  for (int i = 0; i < 8; ++i) { f[i] = lo[i]; f[i + 8] = hh[i]; }
  return f;
}

__global__ __launch_bounds__(256) void k_wmma_gemm(
    const _Float16* __restrict__ A,   // [M][K]
    const _Float16* __restrict__ WT,  // [N][K]  (pre-transposed weights)
    const float* __restrict__ bias, const float* __restrict__ addsrc,
    float* __restrict__ C, int M, int N, int K, int act)
{
  __shared__ _Float16 sA[BMT][LDK];   // [m][k]
  __shared__ _Float16 sB[BNT][LDK];   // [n][k]

  const int tid  = threadIdx.x;
  const int lane = tid & 31;
  const int wave = tid >> 5;
  const int wm   = wave & 3;          // wave M position (4 x 16 rows)
  const int wn   = wave >> 2;         // wave N position (2 x 64 cols)
  const int lr   = lane & 15;
  const int hi   = lane >> 4;
  const int kb   = hi * 8;
  const long bm  = (long)blockIdx.x * BMT;
  const long bn  = (long)blockIdx.y * BNT;

  // cooperative staging: A 2048 halfs (8/thread), B 4096 halfs (16/thread)
  const int ar = tid >> 2, ac = (tid & 3) * 8;
  const int br = tid >> 1, bc = (tid & 1) * 16;

  v8f acc[4] = {v8f{}, v8f{}, v8f{}, v8f{}};

  for (int k0 = 0; k0 < K; k0 += BKT) {
    lds_copy16(A  + (bm + ar) * (long)K + k0 + ac,     &sA[ar][ac]);
    lds_copy16(WT + (bn + br) * (long)K + k0 + bc,     &sB[br][bc]);
    lds_copy16(WT + (bn + br) * (long)K + k0 + bc + 8, &sB[br][bc + 8]);

    if (k0 + BKT < K)
      __builtin_prefetch(A + (bm + ar) * (long)K + k0 + BKT + ac, 0, 0);

    lds_copy_wait();
    __syncthreads();

    const v16h af = load_frag(&sA[wm * 16 + lr][0], kb);
#pragma unroll
    for (int t = 0; t < 4; ++t) {
      const v16h bf = load_frag(&sB[wn * 64 + t * 16 + lr][0], kb);
      acc[t] = __builtin_amdgcn_wmma_f32_16x16x32_f16(false, af, false, bf,
                                                      (short)0, acc[t],
                                                      false, false);
    }
    __syncthreads();
  }

  // epilogue: C layout VGPR r -> M=r (lanes 0-15), M=r+8 (lanes 16-31); N=lr
#pragma unroll
  for (int t = 0; t < 4; ++t) {
    long gn = bn + wn * 64 + t * 16 + lr;
#pragma unroll
    for (int r = 0; r < 8; ++r) {
      long gm = bm + wm * 16 + hi * 8 + r;
      float v = acc[t][r];
      if (bias)   v += bias[gn];
      if (addsrc) v += addsrc[gm * (long)N + gn];
      if (act == 1)      v = fmaxf(v, 0.0f);
      else if (act == 2) v = 0.5f * v * (1.0f + erff(v * 0.70710678118f));
      C[gm * (long)N + gn] = v;
    }
  }
}

// ---------------------------------------------------------------------------
// Elementwise / small kernels
// ---------------------------------------------------------------------------
__global__ void k_cvt(const float* __restrict__ x, _Float16* __restrict__ y, long n) {
  long i = (long)blockIdx.x * blockDim.x + threadIdx.x;
  if (i < n) y[i] = (_Float16)x[i];
}

// dense weight pack: W[K][N] fp32 -> WT[N][K] f16
__global__ void k_cvt_t(const float* __restrict__ W, _Float16* __restrict__ WT,
                        int K, int N) {
  long i = (long)blockIdx.x * blockDim.x + threadIdx.x;
  if (i >= (long)K * N) return;
  int k = i % K; long n = i / K;
  WT[i] = (_Float16)W[(long)k * N + n];
}

// conv weight pack: W (Cout,Cin,3) fp32 -> WT[co][j*Cin+cin] f16 (N-major)
__global__ void k_pack_convT(const float* __restrict__ W, _Float16* __restrict__ o) {
  long i = (long)blockIdx.x * blockDim.x + threadIdx.x;
  if (i >= (long)TCNC * TCNK) return;
  int col = i % TCNK; long co = i / TCNK;
  int j = col / TCNC; int cin = col % TCNC;
  o[i] = (_Float16)W[(co * (long)TCNC + cin) * 3 + j];
}

__global__ void k_fill(float* p, float v, long n) {
  long i = (long)blockIdx.x * blockDim.x + threadIdx.x;
  if (i < n) p[i] = v;
}

__global__ void k_add_relu(float* y, const float* a, long n) {
  long i = (long)blockIdx.x * blockDim.x + threadIdx.x;
  if (i < n) y[i] = fmaxf(y[i] + a[i], 0.0f);
}

// marks: enc marks [16][48][2] and dec marks [16][48][2]
__global__ void k_marks(const float* hist, const float* fut, float* mE, float* mD) {
  int i = blockIdx.x * blockDim.x + threadIdx.x;
  if (i >= BB * LSEQ * 2) return;
  int f = i & 1; int t = (i >> 1) % LSEQ; int b = i / (LSEQ * 2);
  mE[i] = hist[(((long)b * LSEQ + t) * NODES + 0) * 3 + 1 + f];
  float v;
  if (t < PREDL) v = hist[(((long)b * LSEQ + (t + PREDL)) * NODES + 0) * 3 + 1 + f];
  else           v = fut[(((long)b * PREDL + (t - PREDL)) * NODES + 0) * 3 + 1 + f];
  mD[i] = v;
}

// per-(b,node): instance norm stats + affine, node-major normalized series
__global__ void k_norm(const float* hist, const float* aw, const float* ab,
                       float* meanB, float* stdB, float* xn, float* xmean) {
  int i = blockIdx.x * blockDim.x + threadIdx.x;
  if (i >= BN512) return;
  int b = i >> 5, n = i & 31;
  float s = 0.f, s2 = 0.f;
  for (int t = 0; t < LSEQ; t++) {
    float x = hist[(((long)b * LSEQ + t) * NODES + n) * 3];
    s += x; s2 += x * x;
  }
  float mu = s / LSEQ;
  float var = s2 / LSEQ - mu * mu;
  float sd = sqrtf(var + 1e-5f);
  meanB[i] = mu; stdB[i] = sd;
  float A = aw[0], Bc = ab[0], m2 = 0.f;
  for (int t = 0; t < LSEQ; t++) {
    float v = ((hist[(((long)b * LSEQ + t) * NODES + n) * 3] - mu) / sd) * A + Bc;
    xn[(long)i * LSEQ + t] = v; m2 += v;
  }
  xmean[i] = m2 / LSEQ;
}

// seasonal/trend init per (bn, dec-t)
__global__ void k_seastrend(const float* xn, const float* xmean, float* seas, float* trend) {
  long i = (long)blockIdx.x * blockDim.x + threadIdx.x;
  if (i >= (long)BN512 * LSEQ) return;
  int t = i % LSEQ; long bn = i / LSEQ;
  if (t < PREDL) {
    int p = t + PREDL;
    float m = 0.f;
    for (int j = 0; j < 25; j++) {
      int u = p - 12 + j; u = u < 0 ? 0 : (u > LSEQ - 1 ? LSEQ - 1 : u);
      m += xn[bn * LSEQ + u];
    }
    m *= (1.0f / 25.0f);
    seas[i]  = xn[bn * LSEQ + p] - m;
    trend[i] = m;
  } else {
    seas[i]  = 0.f;
    trend[i] = xmean[bn];
  }
}

// data embedding: circular value conv + time features + pos emb + node emb
__global__ void k_embed(const float* xv, const float* mark, const float* Wv,
                        const float* Wt, const float* node, float* out, int L) {
  long i = (long)blockIdx.x * blockDim.x + threadIdx.x;
  long total = (long)BN512 * L * DM;
  if (i >= total) return;
  int d = i % DM; long r = i / DM; int t = r % L; long bn = r / L;
  long bb = bn / NODES; int nd = bn % NODES;
  const float* xs = xv + bn * L;
  float xm1 = xs[(t - 1 + L) % L], x0 = xs[t], xp1 = xs[(t + 1) % L];
  float v = Wv[d * 3 + 0] * xm1 + Wv[d * 3 + 1] * x0 + Wv[d * 3 + 2] * xp1;
  v += mark[(bb * L + t) * 2 + 0] * Wt[0 * DM + d]
     + mark[(bb * L + t) * 2 + 1] * Wt[1 * DM + d];
  float div = __expf(-(float)(d & ~1) * (9.210340372f / (float)DM));
  v += (d & 1) ? __cosf((float)t * div) : __sinf((float)t * div);
  v += node[(long)nd * DM + d];
  out[i] = v;
}

// time-domain circular autocorrelation, mean over heads/dims
__global__ void k_corr(const float* Q, const float* Kt, float* corr, int L) {
  long i = (long)blockIdx.x * blockDim.x + threadIdx.x;
  if (i >= (long)BN512 * L) return;
  int lag = i % L; long bn = i / L;
  const float* q = Q + bn * L * DM;
  const float* k = Kt + bn * L * DM;
  float s = 0.f;
  for (int t = 0; t < L; t++) {
    const float* qa = q + (long)((t + lag) % L) * DM;
    const float* ka = k + (long)t * DM;
    for (int d = 0; d < DM; d++) s += qa[d] * ka[d];
  }
  corr[i] = s * (1.0f / (float)DM);
}

__global__ void k_topk(const float* corr, float* w, int* delay, int L) {
  int bn = blockIdx.x * blockDim.x + threadIdx.x;
  if (bn >= BN512) return;
  const float* c = corr + (long)bn * L;
  float v0 = -1e30f, v1 = -1e30f, v2 = -1e30f; int i0 = 0, i1 = 0, i2 = 0;
  for (int l = 0; l < L; l++) {
    float v = c[l];
    if (v > v0)      { v2 = v1; i2 = i1; v1 = v0; i1 = i0; v0 = v; i0 = l; }
    else if (v > v1) { v2 = v1; i2 = i1; v1 = v; i1 = l; }
    else if (v > v2) { v2 = v; i2 = l; }
  }
  float e0 = __expf(v0 - v0), e1 = __expf(v1 - v0), e2 = __expf(v2 - v0);
  float s = e0 + e1 + e2;
  w[bn * 3 + 0] = e0 / s; w[bn * 3 + 1] = e1 / s; w[bn * 3 + 2] = e2 / s;
  delay[bn * 3 + 0] = i0; delay[bn * 3 + 1] = i1; delay[bn * 3 + 2] = i2;
}

__global__ void k_agg(const float* V, const float* w, const int* delay,
                      float* out, int L) {
  long i = (long)blockIdx.x * blockDim.x + threadIdx.x;
  if (i >= (long)BN512 * L * DM) return;
  int d = i % DM; long r = i / DM; int t = r % L; long bn = r / L;
  const float* wv = w + bn * 3; const int* dl = delay + bn * 3;
  float s = 0.f;
  for (int k = 0; k < TOPK; k++) {
    int u = (t + dl[k]) % L;
    s += wv[k] * V[(bn * L + u) * DM + d];
  }
  out[i] = s;
}

// [Bn,L,256] -> [(bb*8+h)*32+n][t*32+e]  (TCN layout, row len 1536)
__global__ void k_perm_fwd(const float* v2, float* vt, int L) {
  long i = (long)blockIdx.x * blockDim.x + threadIdx.x;
  if (i >= (long)TCNM * TCNC) return;
  int col = i % TCNC; long r = i / TCNC;
  int n = r & 31; long bh = r >> 5; int h = bh & 7; long bb = bh >> 3;
  int t = col >> 5; int e = col & 31;
  vt[i] = v2[(((bb * NODES + n) * (long)L) + t) * DM + h * HD + e];
}

__global__ void k_perm_bwd(const float* vt, float* o, int L) {
  long i = (long)blockIdx.x * blockDim.x + threadIdx.x;
  if (i >= (long)BN512 * L * DM) return;
  int d = i % DM; long r = i / DM; int t = r % L; long bn = r / L;
  int n = bn & 31; long bb = bn >> 5; int h = d >> 5; int e = d & 31;
  o[i] = vt[(((bb * NH + h) * NODES + n) * (long)TCNC) + t * HD + e];
}

// causal dilated im2col over the node axis (len 32), kernel 3, f16 output
__global__ void k_im2col(const float* x, _Float16* out, int dil) {
  long i = (long)blockIdx.x * blockDim.x + threadIdx.x;
  if (i >= (long)TCNM * TCNK) return;
  int col = i % TCNK; long r = i / TCNK;
  int n = r & 31; long bh = r >> 5;
  int j = col / TCNC; int cidx = col % TCNC;
  int u = n - 2 * dil + j * dil;
  float v = (u < 0) ? 0.f : x[(bh * NODES + u) * (long)TCNC + cidx];
  out[i] = (_Float16)v;
}

// moving average series decomposition (k=25, replicate pad)
__global__ void k_decomp(const float* x, float* seas, float* trendAcc, int L) {
  long i = (long)blockIdx.x * blockDim.x + threadIdx.x;
  if (i >= (long)BN512 * L * DM) return;
  int d = i % DM; long r = i / DM; int t = r % L; long bn = r / L;
  float m = 0.f;
  for (int j = 0; j < 25; j++) {
    int u = t - 12 + j; u = u < 0 ? 0 : (u > L - 1 ? L - 1 : u);
    m += x[(bn * L + u) * DM + d];
  }
  m *= (1.0f / 25.0f);
  seas[i] = x[i] - m;
  if (trendAcc) trendAcc[i] += m;
}

// layernorm over D (one block per row)
__global__ void k_ln(const float* x, const float* g, const float* b,
                     float* y, long rows) {
  long r = blockIdx.x; if (r >= rows) return;
  int d = threadIdx.x;
  float v = x[r * DM + d];
  __shared__ float sm[DM];
  sm[d] = v; __syncthreads();
  for (int s = 128; s > 0; s >>= 1) { if (d < s) sm[d] += sm[d + s]; __syncthreads(); }
  float mu = sm[0] / DM; __syncthreads();
  float c = v - mu; sm[d] = c * c; __syncthreads();
  for (int s = 128; s > 0; s >>= 1) { if (d < s) sm[d] += sm[d + s]; __syncthreads(); }
  float var = sm[0] / DM;
  y[r * DM + d] = c * rsqrtf(var + 1e-5f) * g[d] + b[d];
}

// subtract time-axis mean (my_layernorm tail); one thread per (bn,d)
__global__ void k_subtmean(float* y, int L) {
  long i = (long)blockIdx.x * blockDim.x + threadIdx.x;
  if (i >= (long)BN512 * DM) return;
  int d = i % DM; long bn = i / DM;
  float m = 0.f;
  for (int t = 0; t < L; t++) m += y[(bn * L + t) * DM + d];
  m /= (float)L;
  for (int t = 0; t < L; t++) y[(bn * L + t) * DM + d] -= m;
}

// circular trend conv (256ch -> 1ch, kernel 3), accumulated into trend
__global__ void k_trendconv(const float* T, const float* Wt, float* trend) {
  long i = (long)blockIdx.x * blockDim.x + threadIdx.x;
  if (i >= (long)BN512 * LSEQ) return;
  int t = i % LSEQ; long bn = i / LSEQ;
  float s = 0.f;
  for (int d = 0; d < DM; d++) {
    float xm1 = T[(bn * LSEQ + ((t + LSEQ - 1) % LSEQ)) * DM + d];
    float x0  = T[(bn * LSEQ + t) * DM + d];
    float xp1 = T[(bn * LSEQ + ((t + 1) % LSEQ)) * DM + d];
    s += Wt[d * 3 + 0] * xm1 + Wt[d * 3 + 1] * x0 + Wt[d * 3 + 2] * xp1;
  }
  trend[i] += s;
}

// final projection 256 -> 1
__global__ void k_proj(const float* x, const float* Wp, const float* bp,
                       float* o, int L) {
  long i = (long)blockIdx.x * blockDim.x + threadIdx.x;
  if (i >= (long)BN512 * L) return;
  int t = i % L; long bn = i / L;
  float s = bp[0];
  for (int d = 0; d < DM; d++) s += x[(bn * L + t) * DM + d] * Wp[d];
  o[i] = s;
}

// assemble output: de-affine, de-normalize, reorder to (B, PRED, NODES, 1)
__global__ void k_final(const float* season, const float* trend,
                        const float* aw, const float* ab,
                        const float* meanB, const float* stdB, float* out) {
  int i = blockIdx.x * blockDim.x + threadIdx.x;
  if (i >= BB * PREDL * NODES) return;
  int n = i & 31; int r = i >> 5; int t = r % PREDL; int b = r / PREDL;
  long bn = (long)b * NODES + n;
  float v = season[bn * LSEQ + PREDL + t] + trend[bn * LSEQ + PREDL + t];
  v = (v - ab[0]) / (aw[0] + 1e-10f);
  v = v * stdB[bn] + meanB[bn];
  out[((long)b * PREDL + t) * NODES + n] = v;
}

// ---------------------------------------------------------------------------
// Host orchestration
// ---------------------------------------------------------------------------
static inline dim3 g1(long n) { return dim3((unsigned)((n + 255) / 256)); }

struct EmbP  { const float *Wt, *Wv, *node; };
struct AutoP { const float *Wk, *Wo, *Wq, *Wv, *bk, *bo, *bq, *bv; };
struct CrossP{ const float *Wo, *Wv, *bo, *bv, *tW1, *tW2, *tb1, *tb2; };
struct MCP   { AutoP a; CrossP c; };

extern "C" void kernel_launch(void* const* d_in, const int* in_sizes, int n_in,
                              void* d_out, int out_size, void* d_ws, size_t ws_size,
                              hipStream_t stream) {
  (void)in_sizes; (void)n_in; (void)out_size; (void)ws_size;
  int pi = 0;
  const float* hist = (const float*)d_in[pi++];
  const float* fut  = (const float*)d_in[pi++];
  pi += 3; // batch_seen, epoch, train
  auto nx = [&]() { return (const float*)d_in[pi++]; };
  // params leaves, JAX tree (sorted dict key) order
  const float* Wproj = nx();
  const float* affine_b = nx();
  const float* affine_w = nx();
  const float* bproj = nx();
  EmbP decEmb; decEmb.Wt = nx(); decEmb.Wv = nx(); decEmb.node = nx();
  const float *decW1, *decW2, *decWtrend; MCP decCross, decSelf;
  decW1 = nx(); decW2 = nx(); decWtrend = nx();
  auto readMC = [&](MCP& m) {
    m.a.Wk = nx(); m.a.Wo = nx(); m.a.Wq = nx(); m.a.Wv = nx();
    m.a.bk = nx(); m.a.bo = nx(); m.a.bq = nx(); m.a.bv = nx();
    m.c.Wo = nx(); m.c.Wv = nx(); m.c.bo = nx(); m.c.bv = nx();
    m.c.tW1 = nx(); m.c.tW2 = nx(); m.c.tb1 = nx(); m.c.tb2 = nx();
  };
  readMC(decCross); readMC(decSelf);
  const float* decnorm_b = nx(); const float* decnorm_g = nx();
  EmbP encEmb; encEmb.Wt = nx(); encEmb.Wv = nx(); encEmb.node = nx();
  const float *encW1[2], *encW2[2]; MCP encMC[2];
  for (int l = 0; l < 2; l++) { encW1[l] = nx(); encW2[l] = nx(); readMC(encMC[l]); }
  const float* encnorm_b = nx(); const float* encnorm_g = nx();

  // ---- workspace carve ----
  char* wsp = (char*)d_ws; size_t off = 0;
  auto take = [&](size_t bytes) -> void* {
    void* p = wsp + off; off = (off + bytes + 255) & ~(size_t)255; return p;
  };
  const long MK = MROW * DM;                       // 6.29M elements
  float* XA  = (float*)take(MK * 4);
  float* XB  = (float*)take(MK * 4);
  float* Y   = (float*)take(MK * 4);
  float* Qb  = (float*)take(MK * 4);
  float* Kb  = (float*)take(MK * 4);
  float* Vb  = (float*)take(MK * 4);
  float* c1  = (float*)take(MK * 4);
  float* Ff  = (float*)take(MROW * DFF * 4);
  float* Tsum= (float*)take(MK * 4);
  float* encOut = (float*)take(MK * 4);
  float* markE = (float*)take(BB * LSEQ * 2 * 4);
  float* markD = (float*)take(BB * LSEQ * 2 * 4);
  float* meanB = (float*)take(BN512 * 4);
  float* stdB  = (float*)take(BN512 * 4);
  float* xmean = (float*)take(BN512 * 4);
  float* xn    = (float*)take(BN512 * LSEQ * 4);
  float* seas0 = (float*)take(BN512 * LSEQ * 4);
  float* trend0= (float*)take(BN512 * LSEQ * 4);
  float* corr  = (float*)take(BN512 * LSEQ * 4);
  float* wbuf  = (float*)take(BN512 * 3 * 4);
  float* season= (float*)take(BN512 * LSEQ * 4);
  int*   delay = (int*)take(BN512 * 3 * 4);
  _Float16* Ah1 = (_Float16*)take(MK * 2);
  _Float16* Ah2 = (_Float16*)take(MK * 2);
  _Float16* Wh  = (_Float16*)take((long)DM * DFF * 2);
  _Float16* Wch = (_Float16*)take((long)TCNK * TCNC * 2);
  _Float16* A3h = (_Float16*)take((long)TCNM * TCNK * 2);
  _Float16* Fh  = (_Float16*)take(MROW * DFF * 2);

  auto gemm = [&](const _Float16* A, const _Float16* WT, const float* bias,
                  const float* addsrc, float* C, long M, long N, long K, int act) {
    dim3 g((unsigned)(M / BMT), (unsigned)(N / BNT));
    k_wmma_gemm<<<g, 256, 0, stream>>>(A, WT, bias, addsrc, C,
                                       (int)M, (int)N, (int)K, act);
  };
  auto cvt = [&](const float* x, _Float16* y, long n) {
    k_cvt<<<g1(n), 256, 0, stream>>>(x, y, n);
  };
  auto cvtT = [&](const float* W, _Float16* WT, long K, long N) {
    k_cvt_t<<<g1(K * N), 256, 0, stream>>>(W, WT, (int)K, (int)N);
  };

  // multi_correlation: y = xq + auto_corr(xq,xkv) + cross_corr(xkv)
  auto run_mc = [&](const MCP& p, const float* xq, const float* xkv, float* y) {
    cvt(xq,  Ah1, MK);
    cvt(xkv, Ah2, MK);
    cvtT(p.a.Wq, Wh, DM, DM);
    gemm(Ah1, Wh, p.a.bq, nullptr, Qb, MROW, DM, DM, 0);
    cvtT(p.a.Wk, Wh, DM, DM);
    gemm(Ah2, Wh, p.a.bk, nullptr, Kb, MROW, DM, DM, 0);
    cvtT(p.a.Wv, Wh, DM, DM);
    gemm(Ah2, Wh, p.a.bv, nullptr, Vb, MROW, DM, DM, 0);
    k_corr<<<g1((long)BN512 * LSEQ), 256, 0, stream>>>(Qb, Kb, corr, LSEQ);
    k_topk<<<g1(BN512), 256, 0, stream>>>(corr, wbuf, delay, LSEQ);
    k_agg<<<g1(MK), 256, 0, stream>>>(Vb, wbuf, delay, Kb, LSEQ);  // agg -> Kb
    cvt(Kb, Ah1, MK);
    cvtT(p.a.Wo, Wh, DM, DM);
    gemm(Ah1, Wh, p.a.bo, /*addsrc=*/xq, y, MROW, DM, DM, 0);      // y = x + out_t
    // cross branch (TCN over node axis)
    cvtT(p.c.Wv, Wh, DM, DM);
    gemm(Ah2, Wh, p.c.bv, nullptr, Vb, MROW, DM, DM, 0);
    k_perm_fwd<<<g1(MK), 256, 0, stream>>>(Vb, Qb, LSEQ);          // vt -> Qb
    k_im2col<<<g1((long)TCNM * TCNK), 256, 0, stream>>>(Qb, A3h, 1);
    k_pack_convT<<<g1((long)TCNK * TCNC), 256, 0, stream>>>(p.c.tW1, Wch);
    gemm(A3h, Wch, p.c.tb1, nullptr, c1, TCNM, TCNC, TCNK, 1);
    k_im2col<<<g1((long)TCNM * TCNK), 256, 0, stream>>>(c1, A3h, 1);
    k_pack_convT<<<g1((long)TCNK * TCNC), 256, 0, stream>>>(p.c.tW2, Wch);
    gemm(A3h, Wch, p.c.tb2, nullptr, c1, TCNM, TCNC, TCNK, 1);
    k_add_relu<<<g1(MK), 256, 0, stream>>>(c1, Qb, MK);            // residual
    k_perm_bwd<<<g1(MK), 256, 0, stream>>>(c1, Vb, LSEQ);
    cvt(Vb, Ah1, MK);
    cvtT(p.c.Wo, Wh, DM, DM);
    gemm(Ah1, Wh, p.c.bo, /*addsrc=*/y, y, MROW, DM, DM, 0);       // y += out_s
  };

  auto run_ffn = [&](const float* x, const float* W1, const float* W2, float* y) {
    cvt(x, Ah1, MK);
    cvtT(W1, Wh, DM, DFF);
    gemm(Ah1, Wh, nullptr, nullptr, Ff, MROW, DFF, DM, 2);         // gelu
    cvt(Ff, Fh, MROW * DFF);
    cvtT(W2, Wh, DFF, DM);
    gemm(Fh, Wh, nullptr, /*addsrc=*/x, y, MROW, DM, DFF, 0);
  };

  // ---- prep ----
  k_marks<<<g1(BB * LSEQ * 2), 256, 0, stream>>>(hist, fut, markE, markD);
  k_norm<<<g1(BN512), 256, 0, stream>>>(hist, affine_w, affine_b,
                                        meanB, stdB, xn, xmean);
  k_seastrend<<<g1((long)BN512 * LSEQ), 256, 0, stream>>>(xn, xmean, seas0, trend0);

  // ---- encoder ----
  k_embed<<<g1(MK), 256, 0, stream>>>(xn, markE, encEmb.Wv, encEmb.Wt,
                                      encEmb.node, XA, LSEQ);
  float* xcur = XA; float* xalt = XB;
  for (int l = 0; l < 2; l++) {
    run_mc(encMC[l], xcur, xcur, Y);
    k_decomp<<<g1(MK), 256, 0, stream>>>(Y, xalt, nullptr, LSEQ);
    { float* t = xcur; xcur = xalt; xalt = t; }
    run_ffn(xcur, encW1[l], encW2[l], Y);
    k_decomp<<<g1(MK), 256, 0, stream>>>(Y, xalt, nullptr, LSEQ);
    { float* t = xcur; xcur = xalt; xalt = t; }
  }
  k_ln<<<dim3((unsigned)MROW), 256, 0, stream>>>(xcur, encnorm_g, encnorm_b,
                                                 encOut, MROW);
  k_subtmean<<<g1((long)BN512 * DM), 256, 0, stream>>>(encOut, LSEQ);

  // ---- decoder ----
  k_embed<<<g1(MK), 256, 0, stream>>>(seas0, markD, decEmb.Wv, decEmb.Wt,
                                      decEmb.node, XA, LSEQ);
  k_fill<<<g1(MK), 256, 0, stream>>>(Tsum, 0.f, MK);
  run_mc(decSelf, XA, XA, Y);
  k_decomp<<<g1(MK), 256, 0, stream>>>(Y, XB, Tsum, LSEQ);
  run_mc(decCross, XB, encOut, Y);
  k_decomp<<<g1(MK), 256, 0, stream>>>(Y, XA, Tsum, LSEQ);
  run_ffn(XA, decW1, decW2, Y);
  k_decomp<<<g1(MK), 256, 0, stream>>>(Y, XB, Tsum, LSEQ);
  k_trendconv<<<g1((long)BN512 * LSEQ), 256, 0, stream>>>(Tsum, decWtrend, trend0);
  k_ln<<<dim3((unsigned)MROW), 256, 0, stream>>>(XB, decnorm_g, decnorm_b, Y, MROW);
  k_subtmean<<<g1((long)BN512 * DM), 256, 0, stream>>>(Y, LSEQ);
  k_proj<<<g1((long)BN512 * LSEQ), 256, 0, stream>>>(Y, Wproj, bproj, season, LSEQ);

  // ---- output ----
  k_final<<<g1(BB * PREDL * NODES), 256, 0, stream>>>(
      season, trend0, affine_w, affine_b, meanB, stdB, (float*)d_out);
}
